// LSTM_24000277250554
// MI455X (gfx1250) — compile-verified, matrix operand order checked by reading, not run on hardware
//
#include <hip/hip_runtime.h>
#include <hip/hip_bf16.h>
#include <math.h>

// Problem constants (from reference)
#define BB   64      // batch
#define TT   1024    // seq len
#define HH   512     // hidden
#define G4   2048    // 4*H
#define KIN0 64      // layer0 input size
#define OUTD 1024    // fc output
#define NBLK 64      // persistent blocks for LSTM scan (2 tasks x 4 K-splits each)
#define NTHR 256     // 8 waves per block
#define RSTR 40      // padded f32 stride per lane in reduction buffer (mult of 8)

typedef __attribute__((ext_vector_type(16))) _Float16 v16h;
typedef __attribute__((ext_vector_type(8)))  _Float16 v8h;
typedef __attribute__((ext_vector_type(8)))  float    v8f;
typedef __attribute__((ext_vector_type(4)))  int      v4i;

__device__ __forceinline__ v16h cat8(v8h lo, v8h hi) {
  return __builtin_shufflevector(lo, hi, 0,1,2,3,4,5,6,7,8,9,10,11,12,13,14,15);
}
__device__ __forceinline__ float sigm(float x) { return 1.0f / (1.0f + __expf(-x)); }

// ---- CDNA5 async global->LDS copy (16B per lane), tracked by ASYNCcnt ------
__device__ __forceinline__ void async_b128_to_lds(_Float16* dst, const _Float16* src) {
#if __has_builtin(__builtin_amdgcn_global_load_async_to_lds_b128)
  typedef __attribute__((address_space(1))) v4i gv4i;  // global int4
  typedef __attribute__((address_space(3))) v4i lv4i;  // LDS int4
  // int-cast trick: LDS flat addresses carry the LDS byte offset in the low
  // 32 bits, so the truncating int->AS3 cast yields the correct LDS address.
  __builtin_amdgcn_global_load_async_to_lds_b128(
      (gv4i*)(size_t)src, (lv4i*)(size_t)dst, 0, 0);
#else
  unsigned ldsoff = (unsigned)(size_t)dst;   // low 32 bits = LDS byte address
  asm volatile("global_load_async_to_lds_b128 %0, %1, off"
               :: "v"(ldsoff), "v"(src) : "memory");
#endif
}
__device__ __forceinline__ void wait_async_lds() {
#if __has_builtin(__builtin_amdgcn_s_wait_asynccnt)
  __builtin_amdgcn_s_wait_asynccnt(0);
#else
  asm volatile("s_wait_asynccnt 0" ::: "memory");
#endif
}

// ---------------------------------------------------------------- helpers
__global__ void k_cast_f16(_Float16* __restrict__ dst, const float* __restrict__ src, int n) {
  int i = blockIdx.x * blockDim.x + threadIdx.x;
  if (i < n) dst[i] = (_Float16)src[i];
}

__global__ void k_bias_sum(float* __restrict__ dst, const float* __restrict__ a,
                           const float* __restrict__ b, int n) {
  int i = blockIdx.x * blockDim.x + threadIdx.x;
  if (i < n) dst[i] = a[i] + b[i];
}

__global__ void k_zero32(unsigned* __restrict__ p, int nwords) {
  int i = blockIdx.x * blockDim.x + threadIdx.x;
  if (i < nwords) p[i] = 0u;
}

// ---------------------------------------------------------------- layer-1 input projection
// xw[t][gcol][b] = sum_k ula[b][t][k] * wih[gcol][k]   (f16 out, [T][G4][B] layout)
// Full-width GEMM (137 GFLOP) executed once, off the scan's critical path.
// Each wave: one (t, 16-col) strip over all 4 batch tiles -> B fragment reused 4x.
// Transposed output layout makes the scan's per-step consumption 4 contiguous
// v8h loads per lane (accumulator rows are contiguous along B).
__global__ void __launch_bounds__(NTHR)
k_inproj(const _Float16* __restrict__ ula,  // [B][T][H] f16
         const _Float16* __restrict__ wih,  // [G4][H]   f16 (rows = gate cols)
         _Float16* __restrict__ xw)         // [T][G4][B] f16
{
  const int lane = threadIdx.x & 31;
  const int w    = blockIdx.x * (NTHR / 32) + (threadIdx.x >> 5);  // 0..T*128-1
  const int t    = w >> 7;
  const int nc   = w & 127;              // 16-col tile of G4
  const int n    = lane & 15;
  const int kh   = (lane >> 4) << 3;
  const int gcol = nc * 16 + n;

  v8f acc[4] = {};  // one accumulator tile per batch tile
  for (int kb = 0; kb < HH; kb += 32) {
    const _Float16* bp = wih + (size_t)gcol * HH + kb + kh;
    v16h b = cat8(*(const v8h*)bp, *(const v8h*)(bp + 16));
#pragma unroll
    for (int mt = 0; mt < 4; ++mt) {
      const _Float16* ap = ula + ((size_t)(mt * 16 + n) * TT + t) * HH + kb + kh;
      v16h a = cat8(*(const v8h*)ap, *(const v8h*)(ap + 16));
      acc[mt] = __builtin_amdgcn_wmma_f32_16x16x32_f16(
          false, a, false, b, (short)0, acc[mt], false, false);
    }
  }
#pragma unroll
  for (int mt = 0; mt < 4; ++mt) {
    v8h hv;
#pragma unroll
    for (int r = 0; r < 8; ++r) hv[r] = (_Float16)acc[mt][r];
    *(v8h*)&xw[((size_t)t * G4 + gcol) * BB + mt * 16 + kh] = hv;
  }
}

// ---------------------------------------------------------------- LSTM scan
// Persistent-RNN kernel: 64 blocks x 8 waves = 512 waves.
// Block holds 2 tasks; each task (mt = 16 batch rows, jc = 16 hidden cols) is
// split across 4 waves by K. Each wave pins its K=128 slice of W_hh (16 v16h
// fragments = 128 VGPRs) in registers for the entire scan -> zero recurrent
// weight traffic per step. ks=1..3 waves push partial gate tiles through LDS;
// the ks=0 wave reduces, applies the nonlinearities, and owns the f32 cell
// state tile in VGPRs across all T steps. Grid barrier between steps.
// If xw != nullptr the input projection is precomputed ([T][G4][B] f16) and
// folded into the ks0 accumulator init; otherwise it is fused into the scan.
template<int KIN>
__global__ void __launch_bounds__(NTHR)
k_lstm(const _Float16* __restrict__ xh,    // [B][T][KIN] f16
       const _Float16* __restrict__ wih,   // [2048][KIN] f16 (rows = gate cols)
       const _Float16* __restrict__ whh,   // [2048][512] f16 (rows = gate cols)
       const float*    __restrict__ bias,  // [2048] = b_ih + b_hh
       _Float16* __restrict__ hbuf,        // [64][512] f16, zero-initialized
       _Float16* __restrict__ hs_out,      // [B][T][H] f16 or nullptr
       unsigned* __restrict__ bar,         // [0]=arrive, [1]=generation
       const _Float16* __restrict__ xw)    // [T][G4][B] f16 or nullptr
{
  __shared__ _Float16 lds_h[2 * 16 * HH];       // 32 KB: two 16-row h slices
  __shared__ float    red[2 * 3 * 32 * RSTR];   // 30 KB: partial accumulators

  const int lane = threadIdx.x & 31;
  const int wv   = threadIdx.x >> 5;     // wave in block: 0..7
  const int q    = wv >> 2;              // task slot in block: 0/1
  const int ks   = wv & 3;               // K-split id: 0..3
  const int task = blockIdx.x * 2 + q;   // 0..127
  const int mt   = task & 3;             // batch tile
  const int jc   = task >> 2;            // hidden col chunk, 0..31
  const int n    = lane & 15;            // N index / A-row lane map
  const int kh   = (lane >> 4) << 3;     // K half select: 0 or 8
  const int jcol = jc * 16 + n;          // hidden column for this lane
  const int kbase = ks * (HH / 4);       // this wave's K origin: 0,128,256,384

  // ---- pin this wave's W_hh K-slice in registers (once, for all 1024 steps)
  v16h wreg[16];
#pragma unroll
  for (int i = 0; i < 4; ++i) {
#pragma unroll
    for (int g = 0; g < 4; ++g) {
      const _Float16* bp = whh + (size_t)(g * HH + jcol) * HH + kbase + i * 32 + kh;
      wreg[i * 4 + g] = cat8(*(const v8h*)bp, *(const v8h*)(bp + 16));
    }
  }

  float bsv[4];
#pragma unroll
  for (int g = 0; g < 4; ++g) bsv[g] = (ks == 0) ? bias[g * HH + jcol] : 0.0f;

  const _Float16* xlane = xh + (size_t)(mt * 16 + n) * TT * KIN;  // lane's batch row
  const bool lead = (ks == 0);
  v8f cst = {};  // cell state tile (f32), lives in VGPRs of the ks==0 wave

  for (int t = 0; t < TT; ++t) {
    // ---- stage the two needed 16-row h slices via async DMA to LDS
    for (int i = threadIdx.x; i < 2048; i += NTHR) {
      int qq  = i >> 10;
      int c   = i & 1023;
      int mtq = (blockIdx.x * 2 + qq) & 3;
      async_b128_to_lds(&lds_h[qq * 8192 + c * 8], &hbuf[mtq * 16 * HH + c * 8]);
    }
    wait_async_lds();
    __syncthreads();

    // ---- accumulator init: bias (+ precomputed input projection on ks0)
    v8f acc[4];
    if (lead && xw != nullptr) {
#pragma unroll
      for (int g = 0; g < 4; ++g) {
        v8h xv = *(const v8h*)&xw[((size_t)t * G4 + (g * HH + jcol)) * BB + mt * 16 + kh];
#pragma unroll
        for (int r = 0; r < 8; ++r) acc[g][r] = bsv[g] + (float)xv[r];
      }
    } else {
#pragma unroll
      for (int g = 0; g < 4; ++g) {
#pragma unroll
        for (int r = 0; r < 8; ++r) acc[g][r] = bsv[g];
      }
    }

    // ---- recurrent GEMM slice: A from LDS, B from registers (no mem traffic)
#pragma unroll
    for (int i = 0; i < 4; ++i) {
      const _Float16* ap = &lds_h[q * 8192 + n * HH + kbase + i * 32 + kh];
      v16h a = cat8(*(const v8h*)ap, *(const v8h*)(ap + 16));
#pragma unroll
      for (int g = 0; g < 4; ++g) {
        acc[g] = __builtin_amdgcn_wmma_f32_16x16x32_f16(
            false, a, false, wreg[i * 4 + g], (short)0, acc[g], false, false);
      }
    }

    if (xw == nullptr) {
      // ---- fused input projection: K chunks distributed over the 4 ks waves
      const _Float16* xrow = xlane + (size_t)t * KIN;
      constexpr int CH   = KIN / 32;   // 32-wide K chunks
      constexpr int FULL = CH / 4;     // guaranteed rounds per ks wave
#pragma unroll
      for (int u = 0; u < FULL; ++u) {
        const int c = u * 4 + ks;
        const _Float16* ap = xrow + c * 32 + kh;
        v16h a = cat8(*(const v8h*)ap, *(const v8h*)(ap + 16));
#pragma unroll
        for (int g = 0; g < 4; ++g) {
          const _Float16* bp = wih + (size_t)(g * HH + jcol) * KIN + c * 32 + kh;
          v16h b = cat8(*(const v8h*)bp, *(const v8h*)(bp + 16));
          acc[g] = __builtin_amdgcn_wmma_f32_16x16x32_f16(
              false, a, false, b, (short)0, acc[g], false, false);
        }
      }
      if constexpr (CH % 4) {
        if (ks < (CH % 4)) {
          const int c = FULL * 4 + ks;
          const _Float16* ap = xrow + c * 32 + kh;
          v16h a = cat8(*(const v8h*)ap, *(const v8h*)(ap + 16));
#pragma unroll
          for (int g = 0; g < 4; ++g) {
            const _Float16* bp = wih + (size_t)(g * HH + jcol) * KIN + c * 32 + kh;
            v16h b = cat8(*(const v8h*)bp, *(const v8h*)(bp + 16));
            acc[g] = __builtin_amdgcn_wmma_f32_16x16x32_f16(
                false, a, false, b, (short)0, acc[g], false, false);
          }
        }
      }
      if (t + 1 < TT) __builtin_prefetch(xrow + KIN, 0, 0);  // next x_t row
    } else if (lead && t + 1 < TT) {
      // prefetch next step's 4 xw gate tiles (overlaps reduction + barrier)
#pragma unroll
      for (int g = 0; g < 4; ++g) {
        __builtin_prefetch(&xw[((size_t)(t + 1) * G4 + (g * HH + jcol)) * BB + mt * 16], 0, 0);
      }
    }

    // ---- K-split reduction through LDS
    if (!lead) {
      float* dst = &red[(q * 3 + (ks - 1)) * 32 * RSTR + lane * RSTR];
#pragma unroll
      for (int g = 0; g < 4; ++g) *((v8f*)dst + g) = acc[g];
    }
    __syncthreads();

    if (lead) {
#pragma unroll
      for (int s = 0; s < 3; ++s) {
        const float* src = &red[(q * 3 + s) * 32 * RSTR + lane * RSTR];
#pragma unroll
        for (int g = 0; g < 4; ++g) acc[g] += *((const v8f*)src + g);
      }
      // gate nonlinearities + cell update (register elementwise)
      _Float16 hv[8];
#pragma unroll
      for (int r = 0; r < 8; ++r) {
        float iv = sigm(acc[0][r]);
        float fv = sigm(acc[1][r]);
        float gv = tanhf(acc[2][r]);
        float ov = sigm(acc[3][r]);
        float cc = fv * cst[r] + iv * gv;
        cst[r]   = cc;
        hv[r]    = (_Float16)(ov * tanhf(cc));
      }
      // publish h_t (C/D layout: row m = mt*16 + kh + r, col = jcol)
#pragma unroll
      for (int r = 0; r < 8; ++r) {
        int m = mt * 16 + kh + r;
        hbuf[m * HH + jcol] = hv[r];
        if (hs_out) hs_out[((size_t)m * TT + t) * HH + jcol] = hv[r];
      }
    }
    __threadfence();
    __syncthreads();

    // ---- grid-wide barrier (64 co-resident blocks)
    if (threadIdx.x == 0) {
      volatile unsigned* vb = (volatile unsigned*)bar;
      unsigned g0 = vb[1];
      unsigned prev = atomicAdd(bar, 1u);
      if (prev == (unsigned)(gridDim.x - 1)) {
        vb[0] = 0u;
        __threadfence();
        vb[1] = g0 + 1u;
      } else {
        while (vb[1] == g0) { __builtin_amdgcn_s_sleep(2); }
      }
    }
    __syncthreads();
    __threadfence();
  }
}

// ---------------------------------------------------------------- FC + ReLU
// out[r][o] = relu( hcat[r] . fc_w[o] + fc_b[o] ), r in [0,128), o in [0,1024)
__global__ void __launch_bounds__(NTHR)
k_fc(const _Float16* __restrict__ h0,   // [64][512] f16 (layer0 h_T)
     const _Float16* __restrict__ h1,   // [64][512] f16 (layer1 h_T)
     const _Float16* __restrict__ fcw,  // [1024][512] f16 (rows = out cols)
     const float*    __restrict__ fcb,  // [1024] f32
     float* __restrict__ out)           // [128][1024] f32
{
  const int lane = threadIdx.x & 31;
  const int w    = blockIdx.x * (NTHR / 32) + (threadIdx.x >> 5);  // 0..511
  const int mt   = w & 7;    // 8 row tiles (0-3 -> h0, 4-7 -> h1)
  const int nt   = w >> 3;   // 64 col tiles
  const int n    = lane & 15;
  const int kh   = (lane >> 4) << 3;
  const int ncol = nt * 16 + n;
  const _Float16* hsrc = (mt < 4) ? h0 : h1;
  const int arow = (mt & 3) * 16 + n;

  float bv = fcb[ncol];
  v8f acc;
#pragma unroll
  for (int r = 0; r < 8; ++r) acc[r] = bv;

  for (int kb = 0; kb < HH; kb += 32) {
    const _Float16* ap = hsrc + arow * HH + kb + kh;
    v16h a = cat8(*(const v8h*)ap, *(const v8h*)(ap + 16));
    const _Float16* bp = fcw + (size_t)ncol * HH + kb + kh;
    v16h b = cat8(*(const v8h*)bp, *(const v8h*)(bp + 16));
    acc = __builtin_amdgcn_wmma_f32_16x16x32_f16(
        false, a, false, b, (short)0, acc, false, false);
  }
#pragma unroll
  for (int r = 0; r < 8; ++r) {
    int m = mt * 16 + kh + r;
    out[(size_t)m * OUTD + ncol] = fmaxf(acc[r], 0.0f);
  }
}

// ---------------------------------------------------------------- launcher
extern "C" void kernel_launch(void* const* d_in, const int* in_sizes, int n_in,
                              void* d_out, int out_size, void* d_ws, size_t ws_size,
                              hipStream_t stream) {
  (void)in_sizes; (void)n_in; (void)out_size;
  const float* x     = (const float*)d_in[0];
  const float* W_ih0 = (const float*)d_in[1];
  const float* W_hh0 = (const float*)d_in[2];
  const float* b_ih0 = (const float*)d_in[3];
  const float* b_hh0 = (const float*)d_in[4];
  const float* W_ih1 = (const float*)d_in[5];
  const float* W_hh1 = (const float*)d_in[6];
  const float* b_ih1 = (const float*)d_in[7];
  const float* b_hh1 = (const float*)d_in[8];
  const float* fc_w  = (const float*)d_in[9];
  const float* fc_b  = (const float*)d_in[10];
  float* out = (float*)d_out;

  // workspace carve-up (256B aligned)
  char* ws = (char*)d_ws;
  size_t off = 0;
  auto alloc = [&](size_t bytes) -> void* {
    void* p = ws + off;
    off = (off + bytes + 255) & ~(size_t)255;
    return p;
  };
  _Float16* xh     = (_Float16*)alloc((size_t)BB * TT * KIN0 * 2);   // 8 MB
  _Float16* wih0h  = (_Float16*)alloc((size_t)G4 * KIN0 * 2);        // 256 KB
  _Float16* whh0h  = (_Float16*)alloc((size_t)G4 * HH * 2);          // 2 MB
  _Float16* wih1h  = (_Float16*)alloc((size_t)G4 * HH * 2);          // 2 MB
  _Float16* whh1h  = (_Float16*)alloc((size_t)G4 * HH * 2);          // 2 MB
  _Float16* fcwh   = (_Float16*)alloc((size_t)OUTD * HH * 2);        // 1 MB
  float*    bias0  = (float*)alloc((size_t)G4 * 4);
  float*    bias1  = (float*)alloc((size_t)G4 * 4);
  // zero region: h bufs + barrier, contiguous
  _Float16* h0buf  = (_Float16*)alloc((size_t)BB * HH * 2);          // 64 KB
  _Float16* h1buf  = (_Float16*)alloc((size_t)BB * HH * 2);          // 64 KB
  unsigned* bar    = (unsigned*)alloc(256);
  _Float16* ula0   = (_Float16*)alloc((size_t)BB * TT * HH * 2);     // 64 MB
  _Float16* xw1    = (_Float16*)alloc((size_t)TT * G4 * BB * 2);     // 256 MB
  const bool use_xw = (off <= ws_size);   // deterministic per harness

  auto cast = [&](_Float16* dst, const float* src, int n) {
    k_cast_f16<<<(n + NTHR - 1) / NTHR, NTHR, 0, stream>>>(dst, src, n);
  };
  cast(xh,    x,     BB * TT * KIN0);
  cast(wih0h, W_ih0, G4 * KIN0);
  cast(whh0h, W_hh0, G4 * HH);
  cast(wih1h, W_ih1, G4 * HH);
  cast(whh1h, W_hh1, G4 * HH);
  cast(fcwh,  fc_w,  OUTD * HH);
  k_bias_sum<<<(G4 + NTHR - 1) / NTHR, NTHR, 0, stream>>>(bias0, b_ih0, b_hh0, G4);
  k_bias_sum<<<(G4 + NTHR - 1) / NTHR, NTHR, 0, stream>>>(bias1, b_ih1, b_hh1, G4);

  int zwords = (int)(((char*)(bar + 64) - (char*)h0buf) / 4);
  k_zero32<<<(zwords + NTHR - 1) / NTHR, NTHR, 0, stream>>>((unsigned*)h0buf, zwords);

  // layer 0: input x (K=64 fused), writes hidden sequence ula0 + final h0buf
  k_lstm<KIN0><<<NBLK, NTHR, 0, stream>>>(xh, wih0h, whh0h, bias0, h0buf, ula0, bar,
                                          (const _Float16*)nullptr);
  // layer-1 input projection as one full-width GEMM (off the scan critical path)
  if (use_xw) {
    k_inproj<<<TT * 128 / (NTHR / 32), NTHR, 0, stream>>>(ula0, wih1h, xw1);
  }
  // layer 1: recurrence only (xw precomputed), only final h1buf needed
  k_lstm<HH><<<NBLK, NTHR, 0, stream>>>(ula0, wih1h, whh1h, bias1, h1buf,
                                        (_Float16*)nullptr, bar,
                                        use_xw ? (const _Float16*)xw1
                                               : (const _Float16*)nullptr);
  // FC + ReLU on concat(h0_T, h1_T)
  k_fc<<<64, NTHR, 0, stream>>>(h0buf, h1buf, fcwh, fc_b, out);
}
